// SeqNaiveLayer_3564822856011
// MI455X (gfx1250) — matrix operand order repaired and checked by reading, not compiled, not run on hardware
//
#include <hip/hip_runtime.h>

// ---- fragment vector types -------------------------------------------------
typedef __attribute__((ext_vector_type(16))) _Float16 v16h;
typedef __attribute__((ext_vector_type(8)))  _Float16 v8h;
typedef __attribute__((ext_vector_type(4)))  _Float16 v4h;
typedef __attribute__((ext_vector_type(8)))  float    v8f;

// ---- problem constants -----------------------------------------------------
#define S_T   4096
#define B_T   8
#define D_T   1024
#define BN_T  128
#define NTOK  (S_T * B_T)

// tile / LDS layout (strides padded so row stride % 256B == 16B -> no bank conflicts)
#define MT      32            // tokens per workgroup
#define HA_STR  1032          // f16 elems per row of normalized activations
#define SH_STR  1028          // f32 elems per row of shortcut
#define H2_STR  136           // f16 elems per row of GEMM1 output
#define HA_BYTES (MT * HA_STR * 2)   // 66048
#define SH_BYTES (MT * SH_STR * 4)   // 131584
#define H2_BYTES (MT * H2_STR * 2)   // 8704
#define SMEM_BYTES (HA_BYTES + SH_BYTES + H2_BYTES)  // 206336 (< 320KB WGP LDS)

static __device__ __forceinline__ v16h cat8(v8h lo, v8h hi) {
  return __builtin_shufflevector(lo, hi, 0,1,2,3,4,5,6,7,8,9,10,11,12,13,14,15);
}

static __device__ __forceinline__ v8f wmma_f16(v16h a, v16h b, v8f c) {
  // 8 args: (neg_a, A, neg_b, B, c_mod, C, reuse_a, reuse_b)
  return __builtin_amdgcn_wmma_f32_16x16x32_f16(false, a, false, b, (short)0, c,
                                                false, false);
}

// ---- weight fp32 -> fp16 conversion ---------------------------------------
extern "C" __global__ void __launch_bounds__(256)
wcvt_kernel(const float* __restrict__ wd, const float* __restrict__ wu,
            _Float16* __restrict__ wdh, _Float16* __restrict__ wuh, int n) {
  int i = blockIdx.x * blockDim.x + threadIdx.x;
  if (i < n) {
    wdh[i] = (_Float16)wd[i];
    wuh[i] = (_Float16)wu[i];
  }
}

// ---- fused: residual add + LayerNorm + down/ReLU/up + residual -------------
extern "C" __global__ void __launch_bounds__(256)
adapter_fused_kernel(const float* __restrict__ x, const float* __restrict__ res,
                     const float* __restrict__ gamma, const float* __restrict__ beta,
                     const _Float16* __restrict__ WdH, const float* __restrict__ b_down,
                     const _Float16* __restrict__ WuH, const float* __restrict__ b_up,
                     float* __restrict__ out) {
  extern __shared__ char smem[];
  _Float16* hA = (_Float16*)smem;                      // [MT][HA_STR] f16 normalized
  float*    sh = (float*)(smem + HA_BYTES);            // [MT][SH_STR] f32 shortcut
  _Float16* h2 = (_Float16*)(smem + HA_BYTES + SH_BYTES); // [MT][H2_STR] f16 relu(down)

  const int tid  = threadIdx.x;
  const int wave = tid >> 5;
  const int lane = tid & 31;
  const int tok0 = blockIdx.x * MT;

  // ================= Stage 1: residual add + LayerNorm ======================
  // 8 waves x 4 rows each = 32 token rows; each lane handles 32 floats (8x float4)
#pragma unroll
  for (int rr = 0; rr < 4; ++rr) {
    const int m = wave * 4 + rr;
    const int token = tok0 + m;
    const int s = token / B_T;
    const int b = token - s * B_T;
    const float4* xp = (const float4*)(x + (size_t)token * D_T);
    const float4* rp = (const float4*)(res + ((size_t)b * S_T + s) * D_T);

    float4 v[8];
    float sum = 0.f, sq = 0.f;
#pragma unroll
    for (int i = 0; i < 8; ++i) {
      const int d4 = lane + i * 32;          // float4 index; d = 4*d4
      float4 a = xp[d4];
      float4 r = rp[d4];
      float4 t;
      t.x = a.x + r.x; t.y = a.y + r.y; t.z = a.z + r.z; t.w = a.w + r.w;
      v[i] = t;
      sum += t.x + t.y + t.z + t.w;
      sq  += t.x * t.x + t.y * t.y + t.z * t.z + t.w * t.w;
    }
    // wave32 reduction
#pragma unroll
    for (int off = 16; off > 0; off >>= 1) {
      sum += __shfl_xor(sum, off, 32);
      sq  += __shfl_xor(sq,  off, 32);
    }
    const float mu   = sum * (1.0f / D_T);
    const float var  = sq * (1.0f / D_T) - mu * mu;
    const float rstd = rsqrtf(var + 1e-5f);

#pragma unroll
    for (int i = 0; i < 8; ++i) {
      const int d = (lane + i * 32) * 4;
      float4 t = v[i];
      *(float4*)(sh + m * SH_STR + d) = t;           // shortcut (fp32)
      float4 g  = *(const float4*)(gamma + d);
      float4 bb = *(const float4*)(beta + d);
      v4h hv;
      hv[0] = (_Float16)((t.x - mu) * rstd * g.x + bb.x);
      hv[1] = (_Float16)((t.y - mu) * rstd * g.y + bb.y);
      hv[2] = (_Float16)((t.z - mu) * rstd * g.z + bb.z);
      hv[3] = (_Float16)((t.w - mu) * rstd * g.w + bb.w);
      *(v4h*)(hA + m * HA_STR + d) = hv;             // normalized (f16)
    }
  }
  __syncthreads();

  // ================= Stage 2: GEMM1 [MT,D] x Wd^T -> relu -> h2 [MT,BN] =====
  {
    const int col16 = lane & 15;
    const int nT    = wave * 16;                 // BN column tile (8 waves x 16 = 128)
    const int colB  = nT + col16;
    const int koffB = (lane < 16) ? 0 : 16;      // B frag: K-contiguous per lane
    const int kA    = (lane < 16) ? 0 : 16;      // A frag: K-contiguous per lane
    const int rowA0 = col16;
    const int rowA1 = 16 + col16;

    v8f c0 = {}; v8f c1 = {};
#pragma unroll 4
    for (int k = 0; k < D_T; k += 32) {
      const v16h bf = *(const v16h*)(WdH + (size_t)colB * D_T + k + koffB);
      v8h a0lo = *(const v8h*)(hA + rowA0 * HA_STR + k + kA);
      v8h a0hi = *(const v8h*)(hA + rowA0 * HA_STR + k + 8 + kA);
      v8h a1lo = *(const v8h*)(hA + rowA1 * HA_STR + k + kA);
      v8h a1hi = *(const v8h*)(hA + rowA1 * HA_STR + k + 8 + kA);
      c0 = wmma_f16(cat8(a0lo, a0hi), bf, c0);
      c1 = wmma_f16(cat8(a1lo, a1hi), bf, c1);
    }
    const float bd = b_down[colB];
    const int rbase = (lane < 16) ? 0 : 8;
#pragma unroll
    for (int r = 0; r < 8; ++r) {
      const int m0 = r + rbase;                  // C layout: VGPR r -> row r(+8)
      float v0 = c0[r] + bd; v0 = fmaxf(v0, 0.0f);
      h2[m0 * H2_STR + colB] = (_Float16)v0;
      float v1 = c1[r] + bd; v1 = fmaxf(v1, 0.0f);
      h2[(16 + m0) * H2_STR + colB] = (_Float16)v1;
    }
  }
  __syncthreads();

  // ================= Stage 3: GEMM2 [MT,BN] x Wu^T + b_up + shortcut ========
  {
    const int col16 = lane & 15;
    const int kselB = (lane < 16) ? 0 : 16;
    const int kselA = (lane < 16) ? 0 : 16;
    const int rbase = (lane < 16) ? 0 : 8;

#pragma unroll
    for (int nt = 0; nt < 8; ++nt) {
      const int dTile = (nt * 8 + wave) * 16;    // 64 D-tiles over 8 waves
      const int dCol  = dTile + col16;
      v8f c0 = {}; v8f c1 = {};
#pragma unroll
      for (int k = 0; k < BN_T; k += 32) {
        const v16h bf = *(const v16h*)(WuH + (size_t)dCol * BN_T + k + kselB);
        v8h a0lo = *(const v8h*)(h2 + col16 * H2_STR + k + kselA);
        v8h a0hi = *(const v8h*)(h2 + col16 * H2_STR + k + 8 + kselA);
        v8h a1lo = *(const v8h*)(h2 + (16 + col16) * H2_STR + k + kselA);
        v8h a1hi = *(const v8h*)(h2 + (16 + col16) * H2_STR + k + 8 + kselA);
        c0 = wmma_f16(cat8(a0lo, a0hi), bf, c0);
        c1 = wmma_f16(cat8(a1lo, a1hi), bf, c1);
      }
      const float bu = b_up[dCol];
#pragma unroll
      for (int r = 0; r < 8; ++r) {
        const int m0 = r + rbase;
        out[(size_t)(tok0 + m0) * D_T + dCol] =
            c0[r] + bu + sh[m0 * SH_STR + dCol];
        const int m1 = 16 + m0;
        out[(size_t)(tok0 + m1) * D_T + dCol] =
            c1[r] + bu + sh[m1 * SH_STR + dCol];
      }
    }
  }
}

// ---- host launch -----------------------------------------------------------
extern "C" void kernel_launch(void* const* d_in, const int* in_sizes, int n_in,
                              void* d_out, int out_size, void* d_ws, size_t ws_size,
                              hipStream_t stream) {
  const float* x        = (const float*)d_in[0];  // [S,B,D]
  const float* residual = (const float*)d_in[1];  // [B,S,D]
  const float* ln_gamma = (const float*)d_in[2];  // [D]
  const float* ln_beta  = (const float*)d_in[3];  // [D]
  const float* W_down   = (const float*)d_in[4];  // [BN,D]
  const float* b_down   = (const float*)d_in[5];  // [BN]
  const float* W_up     = (const float*)d_in[6];  // [D,BN]
  const float* b_up     = (const float*)d_in[7];  // [D]
  float* out            = (float*)d_out;          // [S,B,D]

  (void)in_sizes; (void)n_in; (void)out_size; (void)ws_size;

  const int nW = BN_T * D_T;                      // 131072 elems each
  _Float16* WdH = (_Float16*)d_ws;
  _Float16* WuH = (_Float16*)((char*)d_ws + (size_t)nW * sizeof(_Float16));

  wcvt_kernel<<<(nW + 255) / 256, 256, 0, stream>>>(W_down, W_up, WdH, WuH, nW);

  adapter_fused_kernel<<<NTOK / MT, 256, SMEM_BYTES, stream>>>(
      x, residual, ln_gamma, ln_beta, WdH, b_down, WuH, b_up, out);
}
